// LiftedStructureLoss_57629871178315
// MI455X (gfx1250) — matrix operand order repaired and checked by reading, not compiled, not run on hardware
//
#include <hip/hip_runtime.h>
#include <hip/hip_bf16.h>
#include <math.h>

// Lifted Structure Loss — N=4096, D=512, 64 classes, margin=1.0
// Plan: bf16 hi/lo split WMMA GEMM for Gram matrix (recomputed in both passes,
// X fits in L2 so recompute is free), deterministic staged reductions.

typedef __attribute__((ext_vector_type(16))) __bf16 v16bf;
typedef __attribute__((ext_vector_type(8)))  float  v8f;

#define N_PTS  4096
#define DIM    512
#define NTILE  (N_PTS / 16)   // 256 row/col tiles
#define MARGIN 1.0f

// ---- fragment loads (ISA 7.12.2 layouts, wave32) ----
// A 16x32 bf16: lane(l) row = l&15; half=l>>4: K = half*8 + {0..7} and 16+half*8 + {0..7}
__device__ __forceinline__ v16bf load_fragA(const __bf16* p /* row base + half*8 + k0 */) {
  union { v16bf v; float4 f[2]; } u;
  u.f[0] = *(const float4*)(p);        // 8 bf16: first K octet
  u.f[1] = *(const float4*)(p + 16);   // 8 bf16: +16 K later
  return u.v;
}
// B 32x16 bf16: lane(l) col = l&15; half=l>>4: K = half*8 + {0..7} and 16+half*8 + {0..7}
__device__ __forceinline__ v16bf load_fragB(const __bf16* p /* row base + half*8 + k0 */) {
  union { v16bf v; float4 f[2]; } u;
  u.f[0] = *(const float4*)(p);
  u.f[1] = *(const float4*)(p + 16);
  return u.v;
}

// 16x16 f32 Gram tile via 3-term bf16 split: G ~= hi*hi + hi*lo + lo*hi
__device__ __forceinline__ v8f gram_tile(const __bf16* __restrict__ Xhi,
                                         const __bf16* __restrict__ Xlo,
                                         int i_base, int j_base, int lane) {
  const int half  = lane >> 4;
  const int idx16 = lane & 15;
  const __bf16* arow_hi = Xhi + (size_t)(i_base + idx16) * DIM + half * 8;
  const __bf16* arow_lo = Xlo + (size_t)(i_base + idx16) * DIM + half * 8;
  const __bf16* brow_hi = Xhi + (size_t)(j_base + idx16) * DIM + half * 8;
  const __bf16* brow_lo = Xlo + (size_t)(j_base + idx16) * DIM + half * 8;
  v8f c = {0.f, 0.f, 0.f, 0.f, 0.f, 0.f, 0.f, 0.f};
#pragma unroll 4
  for (int k0 = 0; k0 < DIM; k0 += 32) {
    v16bf ah = load_fragA(arow_hi + k0);
    v16bf al = load_fragA(arow_lo + k0);
    v16bf bh = load_fragB(brow_hi + k0);
    v16bf bl = load_fragB(brow_lo + k0);
    c = __builtin_amdgcn_wmma_f32_16x16x32_bf16(false, ah, false, bh, (short)0, c, false, false);
    c = __builtin_amdgcn_wmma_f32_16x16x32_bf16(false, ah, false, bl, (short)0, c, false, false);
    c = __builtin_amdgcn_wmma_f32_16x16x32_bf16(false, al, false, bh, (short)0, c, false, false);
  }
  return c;
}

// ---- pass 0: bf16 hi/lo split + exact f32 row norms. One wave per row. ----
__global__ void __launch_bounds__(32)
lsl_prep(const float* __restrict__ X, __bf16* __restrict__ Xhi,
         __bf16* __restrict__ Xlo, float* __restrict__ sq) {
  const int row  = blockIdx.x;
  const int lane = threadIdx.x;              // 0..31, 16 elems each
  const float*  src = X   + (size_t)row * DIM + lane * 16;
  __bf16*       dhi = Xhi + (size_t)row * DIM + lane * 16;
  __bf16*       dlo = Xlo + (size_t)row * DIM + lane * 16;
  float acc = 0.f;
#pragma unroll
  for (int e = 0; e < 16; ++e) {
    float x = src[e];
    __bf16 h = (__bf16)x;
    __bf16 l = (__bf16)(x - (float)h);
    dhi[e] = h;
    dlo[e] = l;
    acc = fmaf(x, x, acc);
  }
#pragma unroll
  for (int m = 1; m <= 16; m <<= 1) acc += __shfl_xor(acc, m, 32);
  if (lane == 0) sq[row] = acc;
}

// ---- pass 1: neg_sum[i] = sum_{j: label!=} exp(margin - d_ij) ----
// 256 blocks (one per 16-row strip) x 4 waves; wave w handles j-tiles w,w+4,...
__global__ void __launch_bounds__(128)
lsl_pass1(const __bf16* __restrict__ Xhi, const __bf16* __restrict__ Xlo,
          const float* __restrict__ sq, const int* __restrict__ labels,
          float* __restrict__ negsum) {
  __shared__ float sred[4][16];
  const int i_base = blockIdx.x * 16;
  const int wave = threadIdx.x >> 5;
  const int lane = threadIdx.x & 31;
  const int half = lane >> 4;
  const int n    = lane & 15;

  float sqi[8]; int li[8];
#pragma unroll
  for (int r = 0; r < 8; ++r) {          // C layout: vgpr r -> row m = r + 8*half
    int i = i_base + half * 8 + r;
    sqi[r] = sq[i];
    li[r]  = labels[i];
  }

  float partial[8] = {0.f, 0.f, 0.f, 0.f, 0.f, 0.f, 0.f, 0.f};
  for (int jt = wave; jt < NTILE; jt += 4) {
    const int j_base = jt * 16;
    v8f c = gram_tile(Xhi, Xlo, i_base, j_base, lane);
    const int   j   = j_base + n;
    const float sqj = sq[j];
    const int   lj  = labels[j];
#pragma unroll
    for (int r = 0; r < 8; ++r) {
      float d2   = fmaxf(sqi[r] + sqj - 2.0f * c[r], 0.0f);
      float dist = sqrtf(d2);
      if (li[r] != lj) partial[r] += expf(MARGIN - dist);
    }
  }
  // sum over the 16 columns held by this half's lanes (butterfly stays in-group)
#pragma unroll
  for (int r = 0; r < 8; ++r) {
    float v = partial[r];
#pragma unroll
    for (int m = 1; m <= 8; m <<= 1) v += __shfl_xor(v, m, 32);
    partial[r] = v;
  }
  if (n == 0) {
#pragma unroll
    for (int r = 0; r < 8; ++r) sred[wave][half * 8 + r] = partial[r];
  }
  __syncthreads();
  if (threadIdx.x < 16) {
    int m = threadIdx.x;                 // deterministic 4-way combine
    negsum[i_base + m] = sred[0][m] + sred[1][m] + sred[2][m] + sred[3][m];
  }
}

// ---- pass 2: hinge over upper-triangular positive pairs + count ----
__global__ void __launch_bounds__(128)
lsl_pass2(const __bf16* __restrict__ Xhi, const __bf16* __restrict__ Xlo,
          const float* __restrict__ sq, const int* __restrict__ labels,
          const float* __restrict__ negsum,
          float* __restrict__ hpart, float* __restrict__ cpart) {
  __shared__ float sh[4], sc[4];
  const int i_base = blockIdx.x * 16;
  const int wave = threadIdx.x >> 5;
  const int lane = threadIdx.x & 31;
  const int half = lane >> 4;
  const int n    = lane & 15;

  float sqi[8], nsi[8]; int li[8];
#pragma unroll
  for (int r = 0; r < 8; ++r) {
    int i = i_base + half * 8 + r;
    sqi[r] = sq[i];
    nsi[r] = negsum[i];
    li[r]  = labels[i];
  }

  float hacc = 0.f, cacc = 0.f;
  for (int jt = wave; jt < NTILE; jt += 4) {
    const int j_base = jt * 16;
    v8f c = gram_tile(Xhi, Xlo, i_base, j_base, lane);
    const int   j   = j_base + n;
    const float sqj = sq[j];
    const float nsj = negsum[j];
    const int   lj  = labels[j];
#pragma unroll
    for (int r = 0; r < 8; ++r) {
      const int i = i_base + half * 8 + r;
      if (li[r] == lj && i < j) {        // upper-triangular positives (diag excluded)
        float d2   = fmaxf(sqi[r] + sqj - 2.0f * c[r], 0.0f);
        float dist = sqrtf(d2);
        float J    = logf(nsi[r] + nsj) + dist;
        float h    = fmaxf(J, 0.0f);
        hacc += h * h;
        cacc += 2.0f;                    // len_p counts BOTH triangles
      }
    }
  }
#pragma unroll
  for (int m = 1; m <= 16; m <<= 1) {
    hacc += __shfl_xor(hacc, m, 32);
    cacc += __shfl_xor(cacc, m, 32);
  }
  if (lane == 0) { sh[wave] = hacc; sc[wave] = cacc; }
  __syncthreads();
  if (threadIdx.x == 0) {
    hpart[blockIdx.x] = sh[0] + sh[1] + sh[2] + sh[3];
    cpart[blockIdx.x] = sc[0] + sc[1] + sc[2] + sc[3];
  }
}

// ---- final: fixed-order fold of 256 partials -> scalar loss ----
__global__ void __launch_bounds__(32)
lsl_final(const float* __restrict__ hpart, const float* __restrict__ cpart,
          float* __restrict__ out) {
  if (threadIdx.x == 0 && blockIdx.x == 0) {
    float h = 0.f, c = 0.f;
    for (int b = 0; b < NTILE; ++b) { h += hpart[b]; c += cpart[b]; }
    out[0] = h / c;
  }
}

extern "C" void kernel_launch(void* const* d_in, const int* in_sizes, int n_in,
                              void* d_out, int out_size, void* d_ws, size_t ws_size,
                              hipStream_t stream) {
  const float* X      = (const float*)d_in[0];
  const int*   labels = (const int*)d_in[1];
  float*       out    = (float*)d_out;

  // workspace layout (~8.03 MB total)
  char*   ws     = (char*)d_ws;
  __bf16* Xhi    = (__bf16*)ws;                                  // 4 MB
  __bf16* Xlo    = Xhi + (size_t)N_PTS * DIM;                    // 4 MB
  float*  sq     = (float*)(ws + 2ull * N_PTS * DIM * sizeof(__bf16));
  float*  negsum = sq + N_PTS;
  float*  hpart  = negsum + N_PTS;
  float*  cpart  = hpart + NTILE;

  lsl_prep <<<N_PTS, 32, 0, stream>>>(X, Xhi, Xlo, sq);
  lsl_pass1<<<NTILE, 128, 0, stream>>>(Xhi, Xlo, sq, labels, negsum);
  lsl_pass2<<<NTILE, 128, 0, stream>>>(Xhi, Xlo, sq, labels, negsum, hpart, cpart);
  lsl_final<<<1, 32, 0, stream>>>(hpart, cpart, out);
}
